// KGAT_80590766342918
// MI455X (gfx1250) — matrix operand order repaired and checked by reading, not compiled, not run on hardware
//
#include <hip/hip_runtime.h>

typedef __attribute__((ext_vector_type(16))) _Float16 v16h;
typedef __attribute__((ext_vector_type(8)))  float    v8f;
typedef __attribute__((ext_vector_type(4)))  float    v4f;

#define DD        128   // entity embedding dim
#define ATTN      128   // attention hidden dim
#define KNEIGH    5
#define BN_PER_WG 16    // (b,n) rows per workgroup
#define NROWS_WG  (BN_PER_WG * KNEIGH)   // 80 neighbor rows per WG = 5 WMMA tiles

static_assert(sizeof(v16h) == 32, "v16h must be 32 bytes");

// ---------------------------------------------------------------------------
// Kernel 0: fold fc1+fc2 (no nonlinearity between them):
//   v1[d] = sum_a W1[128+d, a] * W2[a]   (neighbor-entity block)
//   v2[d] = sum_a W1[256+d, a] * W2[a]   (relation block)
// The entity block v0, b1@W2 and b2 are constant across k and cancel exactly
// in the softmax (shift invariance), so they are not needed at all.
// Stored as _Float16 so phase-1 B-matrix loads are direct 32B v16h loads.
// ---------------------------------------------------------------------------
__global__ __launch_bounds__(256) void kgat_fold_w(const float* __restrict__ W1,
                                                   const float* __restrict__ W2,
                                                   _Float16* __restrict__ v12) {
    const int t = threadIdx.x;            // 0..255 -> rows 128..383 of W1
    const int f = 128 + t;
    float s = 0.0f;
    #pragma unroll 8
    for (int a = 0; a < ATTN; ++a) s += W1[f * ATTN + a] * W2[a];
    v12[t] = (_Float16)s;                 // [0..127]=v1, [128..255]=v2
}

// Load 16 halfs of the A matrix (16x32 f16 WMMA layout) for one K-chunk.
// Contiguous K order: halfs[0..15]=K d0..d0+15
// (with d0 = chunk*32 + (lane>=16 ? 16 : 0)).
__device__ __forceinline__ v16h load_a_f16(const float* __restrict__ rowp, int d0) {
    float t[16];
    *(v4f*)(t + 0)  = *(const v4f*)(rowp + d0);
    *(v4f*)(t + 4)  = *(const v4f*)(rowp + d0 + 4);
    *(v4f*)(t + 8)  = *(const v4f*)(rowp + d0 + 8);
    *(v4f*)(t + 12) = *(const v4f*)(rowp + d0 + 12);
    v16h a;
    #pragma unroll
    for (int i = 0; i < 16; ++i) a[i] = (_Float16)t[i];
    return a;
}

// Same, but non-temporal (rows read exactly once -> TH=NT, don't pollute caches).
__device__ __forceinline__ v16h load_a_f16_nt(const float* __restrict__ rowp, int d0) {
    float t[16];
    *(v4f*)(t + 0)  = __builtin_nontemporal_load((const v4f*)(rowp + d0));
    *(v4f*)(t + 4)  = __builtin_nontemporal_load((const v4f*)(rowp + d0 + 4));
    *(v4f*)(t + 8)  = __builtin_nontemporal_load((const v4f*)(rowp + d0 + 8));
    *(v4f*)(t + 12) = __builtin_nontemporal_load((const v4f*)(rowp + d0 + 12));
    v16h a;
    #pragma unroll
    for (int i = 0; i < 16; ++i) a[i] = (_Float16)t[i];
    return a;
}

// ---------------------------------------------------------------------------
// Kernel 1: per workgroup, 16 (b,n) rows.
// Phase 1 (waves 0-4): WMMA logits. Wave t owns 16 flattened (b,n,k) rows.
//   D_ne += A_ne @ B(v1 bcast)  ;  D_nr += A_nr @ B(v2 bcast)
//   Every lane holds the same B chunk, so every column of D computes the same
//   dot product; we read only column 0 (lanes 0 and 16 of the f32 accumulator).
//   This avoids exec-mask branches entirely. 8 v_wmma per wave, two independent
//   accumulator chains for pipelining.
// Waves 5-7 prefetch the entity lines needed in phase 2 (global_prefetch_b8).
// Phase 2 (all 8 waves, 2 rows each): softmax over K=5 from LDS, coalesced
//   v4f weighted sum of ne (L2-hot re-read), entity copy, NT stores.
// ---------------------------------------------------------------------------
__global__ __launch_bounds__(256) void kgat_main(const float* __restrict__ ent,   // [BN,128]
                                                 const float* __restrict__ ne,    // [BN*5,128]
                                                 const float* __restrict__ nr,    // [BN*5,128]
                                                 const _Float16* __restrict__ v12,
                                                 float* __restrict__ out)         // [BN,256]
{
    __shared__ float s_logit[NROWS_WG];

    const int  tid  = threadIdx.x;
    const int  wave = tid >> 5;
    const int  lane = tid & 31;
    const int  half = lane >> 4;      // 0: lanes 0-15, 1: lanes 16-31
    const int  m    = lane & 15;      // row within 16-row tile / column of D
    const long wg   = blockIdx.x;

    // ---------------- Phase 1: logits via WMMA ----------------
    if (wave < 5) {
        const long  row = wg * NROWS_WG + (long)wave * 16 + m;  // flattened (b,n,k)
        const float* neR = ne + row * DD;
        const float* nrR = nr + row * DD;
        const _Float16* v1h = v12;
        const _Float16* v2h = v12 + 128;

        v8f acc_ne = {};
        v8f acc_nr = {};
        #pragma unroll
        for (int c = 0; c < 4; ++c) {
            const int d0 = c * 32 + half * 16;

            // Broadcast B chunks: lanes of each half load the same 32B (two distinct 32B fetches).
            const v16h bv1 = *(const v16h*)(v1h + c * 32 + half * 16);
            const v16h bv2 = *(const v16h*)(v2h + c * 32 + half * 16);

            const v16h a_ne = load_a_f16(neR, d0);       // re-read in phase 2 -> RT
            acc_ne = __builtin_amdgcn_wmma_f32_16x16x32_f16(
                         false, a_ne, false, bv1, (short)0, acc_ne, false, false);

            const v16h a_nr = load_a_f16_nt(nrR, d0);    // read once -> NT
            acc_nr = __builtin_amdgcn_wmma_f32_16x16x32_f16(
                         false, a_nr, false, bv2, (short)0, acc_nr, false, false);
        }

        // Column 0 of D: lane 0 holds M=0..7, lane 16 holds M=8..15.
        if (m == 0) {
            #pragma unroll
            for (int i = 0; i < 8; ++i)
                s_logit[wave * 16 + half * 8 + i] = acc_ne[i] + acc_nr[i];
        }
    } else {
        // Idle waves: prefetch the 64 x 128B entity lines this WG copies in phase 2.
        const int t2 = (wave - 5) * 32 + lane;           // 0..95
        if (t2 < 64) {
            __builtin_prefetch(ent + wg * BN_PER_WG * DD + t2 * 32, 0, 3);
        }
    }
    __syncthreads();

    // ---------------- Phase 2: softmax + weighted sum ----------------
    #pragma unroll
    for (int r = 0; r < 2; ++r) {
        const int  lw = wave * 2 + r;                  // 0..15 local (b,n)
        const long bn = wg * BN_PER_WG + lw;

        float lg[KNEIGH];
        #pragma unroll
        for (int k = 0; k < KNEIGH; ++k) lg[k] = s_logit[lw * KNEIGH + k];

        float mx = lg[0];
        #pragma unroll
        for (int k = 1; k < KNEIGH; ++k) mx = fmaxf(mx, lg[k]);

        float e[KNEIGH];
        float s = 0.0f;
        #pragma unroll
        for (int k = 0; k < KNEIGH; ++k) { e[k] = __expf(lg[k] - mx); s += e[k]; }
        const float inv = __frcp_rn(s);

        const int d0 = lane * 4;                       // 32 lanes x v4f = 128 floats
        v4f accv = {0.f, 0.f, 0.f, 0.f};
        #pragma unroll
        for (int k = 0; k < KNEIGH; ++k) {
            const v4f  x = *(const v4f*)(ne + (bn * KNEIGH + k) * DD + d0);
            const float a = e[k] * inv;
            accv.x = fmaf(a, x.x, accv.x);
            accv.y = fmaf(a, x.y, accv.y);
            accv.z = fmaf(a, x.z, accv.z);
            accv.w = fmaf(a, x.w, accv.w);
        }

        const v4f ev = *(const v4f*)(ent + bn * DD + d0);
        // Output is write-once, never re-read -> non-temporal stores.
        __builtin_nontemporal_store(ev,   (v4f*)(out + bn * (2 * DD) + d0));
        __builtin_nontemporal_store(accv, (v4f*)(out + bn * (2 * DD) + DD + d0));
    }
}

extern "C" void kernel_launch(void* const* d_in, const int* in_sizes, int n_in,
                              void* d_out, int out_size, void* d_ws, size_t ws_size,
                              hipStream_t stream) {
    const float* ent = (const float*)d_in[0];   // [B,N,128]
    const float* ne  = (const float*)d_in[1];   // [B,N,5,128]
    const float* nr  = (const float*)d_in[2];   // [B,N,5,128]
    const float* W1  = (const float*)d_in[3];   // [384,128]
    // d_in[4] = b1, d_in[6] = b2: constant across k -> cancel in softmax (exact)
    const float* W2  = (const float*)d_in[5];   // [128,1]
    float*       out = (float*)d_out;

    _Float16* v12 = (_Float16*)d_ws;            // 256 halfs = 512 B scratch

    kgat_fold_w<<<1, 256, 0, stream>>>(W1, W2, v12);

    const int BN = in_sizes[0] / DD;            // 1024*50 = 51200
    kgat_main<<<BN / BN_PER_WG, 256, 0, stream>>>(ent, ne, nr, v12, out);
}
